// MoELayer_25331717112093
// MI455X (gfx1250) — matrix operand order repaired and checked by reading, not compiled, not run on hardware
//
#include <hip/hip_runtime.h>
#include <hip/hip_bf16.h>

// ---------------------------------------------------------------------------
// MoE fused kernel for MI455X (gfx1250):
//  - bf16 WMMA (v_wmma_f32_16x16x32_bf16), f32 accumulate
//  - async global->LDS DMA (global_load_async_to_lds_b128 / s_wait_asynccnt)
//    double-buffered weight slices staged once per 64-row block (L2 traffic /4)
//  - flash-style online softmax over experts, accumulator in WMMA D layout
// ---------------------------------------------------------------------------

typedef __attribute__((ext_vector_type(16))) __bf16 v16bf;
typedef __attribute__((ext_vector_type(8)))  float  v8f;

#define B_SZ   32768
#define D_IN   256
#define H1_SZ  256
#define D_OUT  64
#define E_SZ   32
#define MU_F   0.01f

// padded LDS row strides (elements); all rows 16B-aligned
#define LDH   264   // 256-wide bf16 rows (528 B)
#define LDO   72    // 64-wide bf16 rows  (144 B)
#define LDQ   68    // 64-wide f32 rows   (272 B)
#define LDW   40    // 32-wide staged weight slice rows (80 B)
#define LDKV  72    // 64-wide staged K/V rows (144 B)

// dynamic-LDS layout (byte offsets), 64 rows per block
#define XS_OFF   0u                         // 64*264*2 = 33792
#define HS_OFF   33792u                     // 64*264*2 = 33792
#define OS_OFF   67584u                     // 64*72*2  = 9216
#define QS_OFF   76800u                     // 64*68*4  = 17408
#define WB0_OFF  94208u                     // 256*40*2 = 20480
#define WB1_OFF  114688u                    // 256*40*2 = 20480
#define KV_OFF   135168u                    // 2*64*72*2 = 18432  (K then V)
#define KV_VOFF  9216u
#define SC_OFF   153600u                    // 64*4
#define CB_OFF   153856u
#define PB_OFF   154112u
#define MR_OFF   154368u
#define SR_OFF   154624u
#define SMEM_BYTES 154880u

struct U32x8 { uint4 a; uint4 b; };

static __device__ __forceinline__ v16bf make_frag(uint4 lo, uint4 hi) {
  U32x8 t; t.a = lo; t.b = hi;
  return __builtin_bit_cast(v16bf, t);
}

static __device__ __forceinline__ unsigned short f2bf(float f) {
  __bf16 h = (__bf16)f;
  return __builtin_bit_cast(unsigned short, h);
}

static __device__ __forceinline__ v8f wmma_bf16(v16bf a, v16bf b, v8f c) {
  return __builtin_amdgcn_wmma_f32_16x16x32_bf16(
      false, a, false, b, (short)0, c, false, false);
}

// A fragment (16x32 bf16): lane = M row; K chunks [k0+8h, k0+8h+8) and +16
static __device__ __forceinline__ v16bf load_fragA_lds(const unsigned short* lds,
                                                       int row, int ld, int k0, int lane) {
  int kA = k0 + ((lane >> 4) << 3);
  const unsigned short* p = lds + row * ld + kA;
  uint4 lo = *reinterpret_cast<const uint4*>(p);
  uint4 hi = *reinterpret_cast<const uint4*>(p + 16);
  return make_frag(lo, hi);
}

// B fragment (32x16 bf16): lane = N row; K chunks [k0+8h, k0+8h+8) and +16
static __device__ __forceinline__ v16bf load_fragB_lds(const unsigned short* lds,
                                                       int row, int ld, int k0, int lane) {
  int kB = k0 + ((lane >> 4) << 3);
  const unsigned short* p = lds + row * ld + kB;
  uint4 lo = *reinterpret_cast<const uint4*>(p);
  uint4 hi = *reinterpret_cast<const uint4*>(p + 16);
  return make_frag(lo, hi);
}

// ---- CDNA5 async global->LDS DMA (ASYNCcnt-tracked) -----------------------
static __device__ __forceinline__ void async_b128(unsigned lds, const void* g) {
  asm volatile("global_load_async_to_lds_b128 %0, %1, off"
               :: "v"(lds), "v"(g) : "memory");
}
static __device__ __forceinline__ void wait_async0() {
  asm volatile("s_wait_asynccnt 0" ::: "memory");
}

// stage [rows x 32] halves slice (global row stride ldk halves, k offset k0)
// into LDS at lds_base with row stride LDW halves; 256 threads cooperate.
static __device__ __forceinline__ void stage_slice(unsigned lds_base,
                                                   const unsigned short* g,
                                                   int rows, int ldk, int k0, int tid) {
  int chunks = rows * 4;                      // 16B chunks
  for (int c = tid; c < chunks; c += 256) {
    int row = c >> 2, part = c & 3;
    async_b128(lds_base + (unsigned)(row * LDW + part * 8) * 2,
               g + (size_t)row * ldk + k0 + part * 8);
  }
}

// ---------------------------------------------------------------------------
// prepass: f32 -> bf16 copy
__global__ __launch_bounds__(256) void k_cvt_bf16(const float* __restrict__ src,
                                                  unsigned short* __restrict__ dst, int n) {
  int i = blockIdx.x * blockDim.x + threadIdx.x;
  if (i < n) dst[i] = f2bf(src[i]);
}

// prepass: [E,K,N] f32 -> [E,N,K] bf16 (transpose per expert)
__global__ __launch_bounds__(256) void k_transpose_bf16(const float* __restrict__ src,
                                                        unsigned short* __restrict__ dst,
                                                        int E, int K, int N) {
  int i = blockIdx.x * blockDim.x + threadIdx.x;
  int tot = E * K * N;
  if (i >= tot) return;
  int e = i / (K * N);
  int r = i - e * K * N;
  int k = r / N;
  int n = r - k * N;
  dst[(e * N + n) * K + k] = f2bf(src[i]);
}

// regularizer: softmax weights sum to 1 per row => closed form
__global__ void k_regloss(float* __restrict__ out) {
  if (threadIdx.x == 0 && blockIdx.x == 0) {
    float s = (float)B_SZ + (float)B_SZ * (float)E_SZ * 1e-6f;
    out[0] = -(MU_F / (float)E_SZ) * s;
  }
}

// ---------------------------------------------------------------------------
// fused MoE: one block = 8 waves = 64 batch rows
// wave w: m-tile mt = w>>1 (rows mt*16..+16), N-half nh = w&1
__global__ __launch_bounds__(256) void k_moe_fused(
    const unsigned short* __restrict__ Xb,   // [B, 256] bf16
    const unsigned short* __restrict__ W1t,  // [E, 256(n), 256(k)] bf16
    const unsigned short* __restrict__ W2t,  // [E, 64(n), 256(k)]  bf16
    const unsigned short* __restrict__ Kt,   // [E, 64(n), 64(k)]   bf16
    const unsigned short* __restrict__ Vt,   // [E, 64(n), 64(k)]   bf16
    const float* __restrict__ b1,            // [E, 256]
    const float* __restrict__ b2,            // [E, 64]
    const float* __restrict__ tq,            // [T, 64]
    const int*   __restrict__ task,          // [B]
    float* __restrict__ out)                 // [B, 64]
{
  extern __shared__ __align__(16) unsigned char smem[];
  unsigned short* Xs  = (unsigned short*)(smem + XS_OFF);
  unsigned short* hs  = (unsigned short*)(smem + HS_OFF);
  unsigned short* os  = (unsigned short*)(smem + OS_OFF);
  float* qs   = (float*)(smem + QS_OFF);
  float* sc   = (float*)(smem + SC_OFF);
  float* cb   = (float*)(smem + CB_OFF);
  float* pb   = (float*)(smem + PB_OFF);
  float* mrun = (float*)(smem + MR_OFF);
  float* srun = (float*)(smem + SR_OFF);
  const unsigned lds0 = (unsigned)(uintptr_t)smem;   // LDS byte base for DMA

  const int tid  = threadIdx.x;
  const int lane = tid & 31;
  const int wave = tid >> 5;
  const int nl   = lane & 15;
  const int mb   = (lane >> 4) * 8;          // D-fragment row base
  const int mt   = wave >> 1;                // m-tile (0..3)
  const int nh   = wave & 1;                 // N half
  const int r0   = blockIdx.x * 64;
  const int mrow = mt * 16 + nl;             // this lane's LDS A row

  // async stage X tile (64 x 256 bf16)
  for (int c = tid; c < 64 * 32; c += 256) {
    int row = c >> 5, seg = c & 31;
    async_b128(lds0 + XS_OFF + (unsigned)(row * LDH + seg * 8) * 2,
               Xb + (size_t)(r0 + row) * D_IN + seg * 8);
  }
  // per-row task queries (gather -> manual)
  for (int idx = tid; idx < 64 * 16; idx += 256) {
    int row = idx >> 4, c4 = idx & 15;
    int t = task[r0 + row];
    float4 v = *reinterpret_cast<const float4*>(tq + t * D_OUT + c4 * 4);
    *reinterpret_cast<float4*>(&qs[row * LDQ + c4 * 4]) = v;
  }
  if (tid < 64) { mrun[tid] = -3.0e38f; srun[tid] = 0.0f; sc[tid] = 0.0f; }
  wait_async0();
  __syncthreads();

  const v8f vz = {0.f,0.f,0.f,0.f,0.f,0.f,0.f,0.f};
  v8f acc[2] = {vz, vz};                     // flash value accumulators
  const unsigned short* kvK = (const unsigned short*)(smem + KV_OFF);
  const unsigned short* kvV = (const unsigned short*)(smem + KV_OFF + KV_VOFF);

  for (int e = 0; e < E_SZ; ++e) {
    const unsigned short* w1e = W1t + (size_t)e * H1_SZ * D_IN;
    const unsigned short* w2e = W2t + (size_t)e * D_OUT * H1_SZ;
    const unsigned short* ke  = Kt  + (size_t)e * D_OUT * D_OUT;
    const unsigned short* ve  = Vt  + (size_t)e * D_OUT * D_OUT;

    // prefetch K/V for this expert into LDS (completes during GEMM1)
    for (int c = tid; c < 512; c += 256) {
      int row = c >> 3, part = c & 7;
      unsigned off = (unsigned)(row * LDKV + part * 8) * 2;
      async_b128(lds0 + KV_OFF + off,           ke + (size_t)row * D_OUT + part * 8);
      async_b128(lds0 + KV_OFF + KV_VOFF + off, ve + (size_t)row * D_OUT + part * 8);
    }

    // ---- GEMM1: h = relu(X @ W1 + b1); wave covers N in [128*nh, +128) ----
    stage_slice(lds0 + WB0_OFF, w1e, 256, D_IN, 0, tid);
    wait_async0();
    __syncthreads();

    v8f d1[8];
    #pragma unroll
    for (int nt = 0; nt < 8; ++nt) d1[nt] = vz;

    for (int kt = 0; kt < 8; ++kt) {
      const unsigned short* wb =
          (const unsigned short*)(smem + ((kt & 1) ? WB1_OFF : WB0_OFF));
      if (kt < 7)
        stage_slice(lds0 + ((kt & 1) ? WB0_OFF : WB1_OFF),
                    w1e, 256, D_IN, (kt + 1) * 32, tid);
      v16bf a = load_fragA_lds(Xs, mrow, LDH, kt * 32, lane);
      #pragma unroll
      for (int nt = 0; nt < 8; ++nt) {
        int row = nh * 128 + nt * 16 + nl;
        v16bf b = load_fragB_lds(wb, row, LDW, 0, lane);
        d1[nt] = wmma_bf16(a, b, d1[nt]);
      }
      wait_async0();
      __syncthreads();
    }
    #pragma unroll
    for (int nt = 0; nt < 8; ++nt) {
      int col  = nh * 128 + nt * 16 + nl;
      float bi = b1[e * H1_SZ + col];
      #pragma unroll
      for (int i = 0; i < 8; ++i) {
        float f = fmaxf(d1[nt][i] + bi, 0.0f);
        hs[(mt * 16 + mb + i) * LDH + col] = f2bf(f);
      }
    }
    // ---- GEMM2: o = h @ W2 + b2; wave covers N in [32*nh, +32) ----
    stage_slice(lds0 + WB0_OFF, w2e, 64, H1_SZ, 0, tid);
    wait_async0();
    __syncthreads();                          // hs + W2 slice 0 visible

    v8f d2[2] = {vz, vz};
    for (int kt = 0; kt < 8; ++kt) {
      const unsigned short* wb =
          (const unsigned short*)(smem + ((kt & 1) ? WB1_OFF : WB0_OFF));
      if (kt < 7)
        stage_slice(lds0 + ((kt & 1) ? WB0_OFF : WB1_OFF),
                    w2e, 64, H1_SZ, (kt + 1) * 32, tid);
      v16bf a = load_fragA_lds(hs, mrow, LDH, kt * 32, lane);
      #pragma unroll
      for (int nt = 0; nt < 2; ++nt) {
        int row = nh * 32 + nt * 16 + nl;
        v16bf b = load_fragB_lds(wb, row, LDW, 0, lane);
        d2[nt] = wmma_bf16(a, b, d2[nt]);
      }
      wait_async0();
      __syncthreads();
    }
    #pragma unroll
    for (int nt = 0; nt < 2; ++nt) {
      int col  = nh * 32 + nt * 16 + nl;
      float bi = b2[e * D_OUT + col];
      #pragma unroll
      for (int i = 0; i < 8; ++i)
        os[(mt * 16 + mb + i) * LDO + col] = f2bf(d2[nt][i] + bi);
    }
    __syncthreads();                          // os + staged K/V visible

    // ---- GEMM3/4: keys = o @ K[e], values = o @ V[e] ----
    v8f dk[2] = {vz, vz}, dv[2] = {vz, vz};
    #pragma unroll
    for (int kt = 0; kt < 2; ++kt) {
      v16bf a = load_fragA_lds(os, mrow, LDO, kt * 32, lane);
      #pragma unroll
      for (int nt = 0; nt < 2; ++nt) {
        int row = nh * 32 + nt * 16 + nl;
        v16bf bk = load_fragB_lds(kvK, row, LDKV, kt * 32, lane);
        v16bf bv = load_fragB_lds(kvV, row, LDKV, kt * 32, lane);
        dk[nt] = wmma_bf16(a, bk, dk[nt]);
        dv[nt] = wmma_bf16(a, bv, dv[nt]);
      }
    }

    // ---- scores: s[m] = sum_n keys[m][n]*q[m][n] ----
    {
      int c0 = nh * 32 + nl;
      #pragma unroll
      for (int i = 0; i < 8; ++i) {
        int m = mt * 16 + mb + i;
        float p = dk[0][i] * qs[m * LDQ + c0] + dk[1][i] * qs[m * LDQ + c0 + 16];
        p += __shfl_xor(p, 1);
        p += __shfl_xor(p, 2);
        p += __shfl_xor(p, 4);
        p += __shfl_xor(p, 8);
        if (nl == 0) atomicAdd(&sc[m], p);
      }
    }
    __syncthreads();

    // ---- online softmax state update (one thread per row) ----
    if (tid < 64) {
      float s  = sc[tid];
      float mo = mrun[tid];
      float mn = fmaxf(mo, s);
      float c  = __expf(mo - mn);
      float p  = __expf(s - mn);
      srun[tid] = srun[tid] * c + p;
      mrun[tid] = mn;
      cb[tid] = c; pb[tid] = p;
      sc[tid] = 0.0f;
    }
    __syncthreads();

    // ---- rescale accumulator, add weighted values ----
    #pragma unroll
    for (int i = 0; i < 8; ++i) {
      int m = mt * 16 + mb + i;
      float c = cb[m], p = pb[m];
      acc[0][i] = acc[0][i] * c + p * dv[0][i];
      acc[1][i] = acc[1][i] * c + p * dv[1][i];
    }
    __syncthreads();
  }

  // ---- final: tower_input = acc / srun ----
  #pragma unroll
  for (int nt = 0; nt < 2; ++nt) {
    int col = nh * 32 + nt * 16 + nl;
    #pragma unroll
    for (int i = 0; i < 8; ++i) {
      int m = mt * 16 + mb + i;
      out[(size_t)(r0 + m) * D_OUT + col] = acc[nt][i] / srun[m];
    }
  }
}

// ---------------------------------------------------------------------------
extern "C" void kernel_launch(void* const* d_in, const int* in_sizes, int n_in,
                              void* d_out, int out_size, void* d_ws, size_t ws_size,
                              hipStream_t stream) {
  const float* x    = (const float*)d_in[0];
  const int*   task = (const int*)  d_in[1];
  const float* W1   = (const float*)d_in[2];
  const float* b1   = (const float*)d_in[3];
  const float* W2   = (const float*)d_in[4];
  const float* b2   = (const float*)d_in[5];
  const float* tq   = (const float*)d_in[6];
  const float* Km   = (const float*)d_in[7];
  const float* Vm   = (const float*)d_in[8];
  float* out = (float*)d_out;

  // workspace layout (bf16 halves); ~22.5 MB
  unsigned short* Xb  = (unsigned short*)d_ws;
  unsigned short* W1t = Xb  + (size_t)B_SZ * D_IN;
  unsigned short* W2t = W1t + (size_t)E_SZ * H1_SZ * D_IN;
  unsigned short* Kt  = W2t + (size_t)E_SZ * D_OUT * H1_SZ;
  unsigned short* Vt  = Kt  + (size_t)E_SZ * D_OUT * D_OUT;

  {
    int n = B_SZ * D_IN;
    k_cvt_bf16<<<(n + 255) / 256, 256, 0, stream>>>(x, Xb, n);
  }
  {
    int n = E_SZ * D_IN * H1_SZ;
    k_transpose_bf16<<<(n + 255) / 256, 256, 0, stream>>>(W1, W1t, E_SZ, D_IN, H1_SZ);
  }
  {
    int n = E_SZ * H1_SZ * D_OUT;
    k_transpose_bf16<<<(n + 255) / 256, 256, 0, stream>>>(W2, W2t, E_SZ, H1_SZ, D_OUT);
  }
  {
    int n = E_SZ * D_OUT * D_OUT;
    k_transpose_bf16<<<(n + 255) / 256, 256, 0, stream>>>(Km, Kt, E_SZ, D_OUT, D_OUT);
    k_transpose_bf16<<<(n + 255) / 256, 256, 0, stream>>>(Vm, Vt, E_SZ, D_OUT, D_OUT);
  }

  // fused MoE: 512 blocks x 256 threads (8 waves, 64 rows per block)
  k_moe_fused<<<B_SZ / 64, 256, SMEM_BYTES, stream>>>(Xb, W1t, W2t, Kt, Vt,
                                                      b1, b2, tq, task, out);

  k_regloss<<<1, 32, 0, stream>>>(out + (size_t)B_SZ * D_OUT);
}
